// GRUNet_6038724018316
// MI455X (gfx1250) — matrix operand order repaired and checked by reading, not compiled, not run on hardware
//
#include <hip/hip_runtime.h>

#define B_ 256
#define T_ 2048
#define H_ 64
#define LDS_STRIDE 72   // f16 elems per batch row: 64 + pad -> 144B row, keeps 16B alignment

typedef __attribute__((ext_vector_type(16))) _Float16 v16h;
typedef __attribute__((ext_vector_type(8)))  _Float16 v8h;
typedef __attribute__((ext_vector_type(8)))  float    v8f;

__device__ __forceinline__ float fast_sig(float x) {
  // sigmoid via native v_exp_f32 (exp2) + v_rcp_f32
  float e = __builtin_amdgcn_exp2f(x * -1.442695040888963f);
  return __builtin_amdgcn_rcpf(1.0f + e);
}
__device__ __forceinline__ float fast_tanh(float x) {
  return 2.0f * fast_sig(2.0f * x) - 1.0f;
}

__global__ __launch_bounds__(128) void gru_wmma_kernel(
    const float* __restrict__ x,    const float* __restrict__ W_ih,
    const float* __restrict__ W_hh, const float* __restrict__ b_ih,
    const float* __restrict__ b_hh, float* __restrict__ out)
{
  // Ping-pong h buffers: [batch-in-tile][hidden], f16, padded rows (16B aligned).
  __shared__ __align__(16) _Float16 hbuf[2][16 * LDS_STRIDE];

  const int lane = threadIdx.x & 31;
  const int wave = threadIdx.x >> 5;      // 0..3 -> 16-wide hidden slice
  const int ln   = lane & 15;
  const int hi   = lane >> 4;             // 0: lanes 0-15, 1: lanes 16-31
  const int batchBase = blockIdx.x * 16;
  const int hidBase   = wave * 16;

  // ---- W_hh rows as resident WMMA A-fragments (f32 -> packed f16).
  // A 16x32 f16 layout: lanes<16 row M=ln hold K = {c*32+0..7, c*32+16..23};
  //                     lanes>=16 same row hold K = {c*32+8..15, c*32+24..31}.
  v16h A[3][2];
#pragma unroll
  for (int g = 0; g < 3; ++g) {
    const int row = g * H_ + hidBase + ln;
#pragma unroll
    for (int c = 0; c < 2; ++c) {
      const int k0 = c * 32 + hi * 8;
      const float* wr = W_hh + row * H_ + k0;
      float tmp[16];
      *(float4*)(tmp + 0)  = ((const float4*)(wr))[0];
      *(float4*)(tmp + 4)  = ((const float4*)(wr))[1];
      *(float4*)(tmp + 8)  = ((const float4*)(wr + 16))[0];
      *(float4*)(tmp + 12) = ((const float4*)(wr + 16))[1];
#pragma unroll
      for (int e = 0; e < 16; ++e) A[g][c][e] = (_Float16)tmp[e];
    }
  }

  // ---- Per-row gate constants in D layout (VGPR v -> gate row g*64+hidBase+8*hi+v).
  float wih[3][8];       // input weight per gate row (I == 1)
  float sumrz[2][8];     // b_hh + b_ih for r,z
  float bhh_n[8], bih_n[8];
#pragma unroll
  for (int g = 0; g < 3; ++g) {
    const int rb = g * H_ + hidBase + 8 * hi;
    float bh[8], bi[8], wi[8];
    *(float4*)(bh + 0) = ((const float4*)(b_hh + rb))[0];
    *(float4*)(bh + 4) = ((const float4*)(b_hh + rb))[1];
    *(float4*)(bi + 0) = ((const float4*)(b_ih + rb))[0];
    *(float4*)(bi + 4) = ((const float4*)(b_ih + rb))[1];
    *(float4*)(wi + 0) = ((const float4*)(W_ih + rb))[0];
    *(float4*)(wi + 4) = ((const float4*)(W_ih + rb))[1];
#pragma unroll
    for (int v = 0; v < 8; ++v) {
      wih[g][v] = wi[v];
      if (g == 0) sumrz[0][v] = bh[v] + bi[v];
      if (g == 1) sumrz[1][v] = bh[v] + bi[v];
      if (g == 2) { bhh_n[v] = bh[v]; bih_n[v] = bi[v]; }
    }
  }

  const float* xp   = x   + (size_t)(batchBase + ln) * T_;
  float*       outp = out + (size_t)(batchBase + ln) * T_;
  const bool writer = (wave == 3) && (hi == 1);   // hidden channel 63 lives here (v=7)

  // ---- Recurrent state: h as B-fragments (h0 = 0) + this wave's D-layout slice.
  v16h Bv0 = {}, Bv1 = {};
  float hstate[8];
#pragma unroll
  for (int v = 0; v < 8; ++v) hstate[v] = 0.0f;

  // One GRU step. parity selects the LDS ping-pong buffer (compile-time after unroll).
  auto step = [&](const int parity, const float xv, float& outv) {
    // G = W_hh(slice) * h^T : 3 gate tiles, K=64 split as 2 x K32.
    v8f accR = {}, accZ = {}, accN = {};
    accR = __builtin_amdgcn_wmma_f32_16x16x32_f16(false, A[0][0], false, Bv0, (short)0, accR, false, false);
    accR = __builtin_amdgcn_wmma_f32_16x16x32_f16(false, A[0][1], false, Bv1, (short)0, accR, false, false);
    accZ = __builtin_amdgcn_wmma_f32_16x16x32_f16(false, A[1][0], false, Bv0, (short)0, accZ, false, false);
    accZ = __builtin_amdgcn_wmma_f32_16x16x32_f16(false, A[1][1], false, Bv1, (short)0, accZ, false, false);
    accN = __builtin_amdgcn_wmma_f32_16x16x32_f16(false, A[2][0], false, Bv0, (short)0, accN, false, false);
    accN = __builtin_amdgcn_wmma_f32_16x16x32_f16(false, A[2][1], false, Bv1, (short)0, accN, false, false);

    v8h hpk;
#pragma unroll
    for (int v = 0; v < 8; ++v) {
      float r  = fast_sig(accR[v] + sumrz[0][v] + xv * wih[0][v]);
      float z  = fast_sig(accZ[v] + sumrz[1][v] + xv * wih[1][v]);
      float hn = accN[v] + bhh_n[v];
      float xn = xv * wih[2][v] + bih_n[v];
      float n  = fast_tanh(xn + r * hn);
      float h  = n + z * (hstate[v] - n);   // (1-z)*n + z*h
      hstate[v] = h;
      hpk[v] = (_Float16)h;
    }
    outv = hstate[7];                       // h_new[:, -1] for this timestep

    // Publish h_new slice: lane = batch col, contiguous 8 f16 -> one ds_store_b128.
    *(v8h*)(&hbuf[parity][ln * LDS_STRIDE + hidBase + hi * 8]) = hpk;
    __syncthreads();

    // Reload full h as B-fragments for next step:
    // lanes<16: N=ln, K={c*32+0..7, c*32+16..23} ; lanes>=16: N=ln, K={c*32+8..15, c*32+24..31}.
    const _Float16* src = &hbuf[parity][ln * LDS_STRIDE + hi * 8];
    const v8h b0 = *(const v8h*)(src);
    const v8h b1 = *(const v8h*)(src + 16);
    const v8h b2 = *(const v8h*)(src + 32);
    const v8h b3 = *(const v8h*)(src + 48);
#pragma unroll
    for (int e = 0; e < 8; ++e) {
      Bv0[e] = b0[e]; Bv0[8 + e] = b1[e];
      Bv1[e] = b2[e]; Bv1[8 + e] = b3[e];
    }
  };

  // Software-pipelined x fetch: one float4 per 4 steps, prefetched a full group
  // (4 barriers + 24 WMMAs) ahead so global latency never lands on the scan path.
  float4 xg = *(const float4*)(xp);
#pragma unroll 1
  for (int t0 = 0; t0 < T_; t0 += 4) {
    const float4 xc = xg;
    const int nt = (t0 + 4 < T_) ? (t0 + 4) : 0;   // wrap: harmless dead prefetch on last group
    xg = *(const float4*)(xp + nt);

    float4 ob;
    step(0, xc.x, ob.x);
    step(1, xc.y, ob.y);
    step(0, xc.z, ob.z);
    step(1, xc.w, ob.w);

    if (writer) *(float4*)(outp + t0) = ob;        // one b128 store per 4 timesteps
  }
}

extern "C" void kernel_launch(void* const* d_in, const int* in_sizes, int n_in,
                              void* d_out, int out_size, void* d_ws, size_t ws_size,
                              hipStream_t stream) {
  const float* x    = (const float*)d_in[0];
  const float* W_ih = (const float*)d_in[1];
  const float* W_hh = (const float*)d_in[2];
  const float* b_ih = (const float*)d_in[3];
  const float* b_hh = (const float*)d_in[4];
  float* out = (float*)d_out;

  dim3 grid(B_ / 16);   // 16 workgroups, one 16-batch tile each
  dim3 block(128);      // 4 waves: one 16-wide hidden slice per wave
  hipLaunchKernelGGL(gru_wmma_kernel, grid, block, 0, stream,
                     x, W_ih, W_hh, b_ih, b_hh, out);
}
